// ConvPixelToCapsules_59442347377103
// MI455X (gfx1250) — matrix operand order repaired and checked by reading, not compiled, not run on hardware
//
#include <hip/hip_runtime.h>
#include <stdint.h>

// ---------------------------------------------------------------------------
// Fused CapsNet ConvPixelToCapsules for gfx1250 (MI455X).
//   Phase 1: implicit-GEMM 3x3 conv via v_wmma_f32_16x16x32_bf16, votes -> LDS
//   Phase 2: 3 dynamic-routing iterations fully in LDS/registers (1 wave/pixel)
// All LDS fragment traffic is b128 (ds_load_b128 / 16B-aligned vectors).
// ---------------------------------------------------------------------------

typedef __attribute__((ext_vector_type(16))) __bf16    v16bf;
typedef __attribute__((ext_vector_type(8)))  float     v8f;
typedef __attribute__((ext_vector_type(4)))  uint32_t  u32x4;
typedef __attribute__((ext_vector_type(4)))  float     f32x4;

#define BS   8
#define CI   32
#define NI   8
#define HI   32
#define WI   32
#define CO_  32
#define NO_  8
#define KDIM 72      // NI * 3 * 3
#define KPAD 96      // padded to 3 chunks of 32 for bf16 WMMA
#define PIX  8       // pixels per workgroup
#define MROWS 256    // PIX * CI
#define NCOLS 256    // CO_ * NO_

union Frag { v16bf v; u32x4 q[2]; };

__device__ __forceinline__ float bflo(uint32_t u) {
    union { uint32_t i; float f; } c; c.i = u << 16; return c.f;
}
__device__ __forceinline__ float bfhi(uint32_t u) {
    union { uint32_t i; float f; } c; c.i = u & 0xffff0000u; return c.f;
}

__global__ __launch_bounds__(256)
void capsule_conv_route_fused(const float* __restrict__ x,
                              const float* __restrict__ w,
                              const float* __restrict__ bias,
                              float* __restrict__ out)
{
    extern __shared__ char smem[];
    // Phase-1 region [0, 96K): A (48K) + Bt (48K).  Reused in phase 2 as
    // route (32K) + act (8K).  Votes live at [96K, 224K) for both phases.
    __bf16*   A_bf  = (__bf16*)smem;                  // [256][96] im2col, bf16
    __bf16*   B_bf  = (__bf16*)(smem + 48 * 1024);    // [256][96] weights (n-major)
    uint32_t* A_u   = (uint32_t*)A_bf;
    uint32_t* B_u   = (uint32_t*)B_bf;
    float*    route = (float*)smem;                    // [8][32][32]
    float*    act   = (float*)(smem + 32 * 1024);      // [8][256]
    __bf16*   V_bf  = (__bf16*)(smem + 96 * 1024);     // [256][256] votes bf16

    const int tid  = threadIdx.x;
    const int wv   = tid >> 5;        // wave id 0..7
    const int lane = tid & 31;
    const int half = lane >> 4;
    const int r    = lane & 15;

    const int blk = blockIdx.x;
    const int b   = blk >> 7;             // batch (8)
    const int rem = blk & 127;
    const int y   = rem >> 2;             // output row (32)
    const int x0  = (rem & 3) * PIX;      // 8-pixel segment start

    // Warm the L2/WGP$ with the three input rows this block consumes.
    for (int t = tid; t < CI * NI * 3; t += 256) {
        const int ci = t / (NI * 3);
        const int ni = (t / 3) % NI;
        int yy = y + (t % 3) - 1;
        yy = (yy < 0) ? 0 : ((yy >= HI) ? HI - 1 : yy);
        __builtin_prefetch(&x[(((b * CI + ci) * NI + ni) * HI + yy) * WI + x0], 0, 0);
    }

    // ---------------- Phase 1a: stage A (im2col) and Bt into LDS -----------
    for (int e = tid; e < MROWS * KPAD; e += 256) {
        const int m = e / KPAD;
        const int k = e - m * KPAD;
        float va = 0.f;
        if (k < KDIM) {
            const int ci = m & 31, p = m >> 5;
            const int ni = k / 9, r9 = k - ni * 9;
            const int ky = r9 / 3, kx = r9 - ky * 3;
            const int yy = y + ky - 1, xx = x0 + p + kx - 1;
            if ((unsigned)yy < HI && (unsigned)xx < WI)
                va = x[(((b * CI + ci) * NI + ni) * HI + yy) * WI + xx];
        }
        A_bf[e] = (__bf16)va;
        const float vb = (k < KDIM) ? w[m * KDIM + k] : 0.f; // m acts as n
        B_bf[e] = (__bf16)vb;
    }
    __syncthreads();

    // ---------------- Phase 1b: 256x256x96 GEMM via WMMA -------------------
    // votes[m][n] = sum_k A[m][k] * Bt[n][k];  m=(pix,ci), n=(co,no)
    for (int mtb = 0; mtb < 2; ++mtb) {
        const int mt = wv + mtb * 8;          // tile row, 16 M-rows
        v8f acc[16];
        const v8f vzero = {0.f, 0.f, 0.f, 0.f, 0.f, 0.f, 0.f, 0.f};
        #pragma unroll
        for (int nt = 0; nt < 16; ++nt) acc[nt] = vzero;

        #pragma unroll
        for (int kc = 0; kc < 3; ++kc) {
            // A fragment (16x32 bf16, ISA 7.12.2 layout), hoisted over nt.
            // Lane reads two 16B-aligned b128 chunks.
            Frag af;
            const u32x4* arow4 =
                (const u32x4*)(A_u + (mt * 16 + r) * (KPAD / 2) + kc * 16 + half * 4);
            af.q[0] = arow4[0];
            af.q[1] = arow4[2];   // +32 bytes: pairs K=16..23 / 24..31

            #pragma unroll
            for (int nt = 0; nt < 16; ++nt) {
                Frag bf;
                const u32x4* brow4 =
                    (const u32x4*)(B_u + (nt * 16 + r) * (KPAD / 2) + kc * 16 + half * 4);
                bf.q[0] = brow4[0];
                bf.q[1] = brow4[2];
                acc[nt] = __builtin_amdgcn_wmma_f32_16x16x32_bf16(
                    false, af.v, false, bf.v, (short)0, acc[nt], false, false);
            }
        }
        // C/D layout: VGPR i -> M = i + 8*half, N = r (within 16x16 tile)
        #pragma unroll
        for (int nt = 0; nt < 16; ++nt) {
            #pragma unroll
            for (int i = 0; i < 8; ++i) {
                const int m = mt * 16 + half * 8 + i;
                const int n = nt * 16 + r;
                V_bf[m * NCOLS + n] = (__bf16)acc[nt][i];
            }
        }
    }
    __syncthreads();

    // ---------------- Phase 2: dynamic routing, one wave per pixel ---------
    const int p = wv;
    const uint32_t* Vci_u = (const uint32_t*)(V_bf + (p * CI + lane) * NCOLS);
    const uint32_t* Vp_u  = (const uint32_t*)(V_bf + p * CI * NCOLS);

    float logits[CO_];
    #pragma unroll
    for (int c = 0; c < CO_; ++c) logits[c] = 0.f;

    float biasr[NO_];
    #pragma unroll
    for (int q = 0; q < NO_; ++q) biasr[q] = bias[lane * NO_ + q]; // lane = co

    float vact[NO_];

    for (int it = 0; it < 3; ++it) {
        // --- softmax over co (lane = ci) ---
        float mx = logits[0];
        #pragma unroll
        for (int c = 1; c < CO_; ++c) mx = fmaxf(mx, logits[c]);
        float ex[CO_], sum = 0.f;
        #pragma unroll
        for (int c = 0; c < CO_; ++c) { ex[c] = __expf(logits[c] - mx); sum += ex[c]; }
        const float inv = 1.f / sum;
        #pragma unroll
        for (int c = 0; c < CO_; ++c)
            route[(p * CI + lane) * CO_ + c] = ex[c] * inv;
        __syncthreads();

        // --- s_j = sum_ci route * votes + bias, then squash (lane = co) ---
        float s[NO_];
        #pragma unroll
        for (int q = 0; q < NO_; ++q) s[q] = biasr[q];
        for (int ci = 0; ci < CI; ++ci) {
            const float rt = route[(p * CI + ci) * CO_ + lane];
            const u32x4 uv = *(const u32x4*)(Vp_u + ci * (NCOLS / 2) + lane * 4);
            #pragma unroll
            for (int j = 0; j < 4; ++j) {
                const uint32_t u = uv[j];
                s[2 * j]     = fmaf(rt, bflo(u), s[2 * j]);
                s[2 * j + 1] = fmaf(rt, bfhi(u), s[2 * j + 1]);
            }
        }
        float n2 = 0.f;
        #pragma unroll
        for (int q = 0; q < NO_; ++q) n2 = fmaf(s[q], s[q], n2);
        const float fac = sqrtf(n2) / (1.f + n2);   // x*||x||/(1+||x||^2)
        f32x4 v0, v1;
        #pragma unroll
        for (int q = 0; q < 4; ++q) { vact[q] = s[q] * fac; v0[q] = vact[q]; }
        #pragma unroll
        for (int q = 0; q < 4; ++q) { vact[4 + q] = s[4 + q] * fac; v1[q] = vact[4 + q]; }
        f32x4* actw = (f32x4*)(act + p * NCOLS + lane * NO_);
        actw[0] = v0;
        actw[1] = v1;
        __syncthreads();

        // --- logits += <votes, v>  (lane = ci) ---
        if (it < 2) {
            #pragma unroll
            for (int c = 0; c < CO_; ++c) {
                const u32x4 uv = *(const u32x4*)(Vci_u + c * 4);
                const f32x4* ar = (const f32x4*)(act + p * NCOLS + c * NO_);
                const f32x4 a0 = ar[0];
                const f32x4 a1 = ar[1];
                float d = 0.f;
                d = fmaf(bflo(uv[0]), a0[0], d);
                d = fmaf(bfhi(uv[0]), a0[1], d);
                d = fmaf(bflo(uv[1]), a0[2], d);
                d = fmaf(bfhi(uv[1]), a0[3], d);
                d = fmaf(bflo(uv[2]), a1[0], d);
                d = fmaf(bfhi(uv[2]), a1[1], d);
                d = fmaf(bflo(uv[3]), a1[2], d);
                d = fmaf(bfhi(uv[3]), a1[3], d);
                logits[c] += d;
            }
            __syncthreads();   // route region is rewritten next iteration
        }
    }

    // --- write activation [bs, co, no, ho, wo]  (lane = co) ---
    const int xp = x0 + p;
    #pragma unroll
    for (int q = 0; q < NO_; ++q)
        out[(((b * CO_ + lane) * NO_ + q) * HI + y) * WI + xp] = vact[q];
}

extern "C" void kernel_launch(void* const* d_in, const int* in_sizes, int n_in,
                              void* d_out, int out_size, void* d_ws, size_t ws_size,
                              hipStream_t stream) {
    (void)in_sizes; (void)n_in; (void)d_ws; (void)ws_size; (void)out_size;
    const float* x    = (const float*)d_in[0];
    const float* w    = (const float*)d_in[1];
    const float* bias = (const float*)d_in[2];
    float* out        = (float*)d_out;

    const size_t shmem = (size_t)224 * 1024;  // 96K staging/route+act, 128K votes
    hipFuncSetAttribute((const void*)capsule_conv_route_fused,
                        hipFuncAttributeMaxDynamicSharedMemorySize, (int)shmem);

    // grid: 8 batches x 32 rows x 4 eight-pixel segments = 1024 workgroups
    capsule_conv_route_fused<<<dim3(BS * HI * (WI / PIX)), dim3(256), shmem, stream>>>(
        x, w, bias, out);
}
